// RITS_1632087572738
// MI455X (gfx1250) — matrix operand order repaired and checked by reading, not compiled, not run on hardware
//
#include <hip/hip_runtime.h>
#include <hip/hip_bf16.h>

#define B_ 4096
#define T_ 64
#define F_ 64
#define H_ 128
#define F2_ 128
#define G_ 512
#define TILE_M 16
#define NBLK (B_ / TILE_M)   /* 256 */
#define NTHREADS 128
#define EPSF 1e-12f

// f16 weight workspace layout (element offsets)
#define WDH_OFF    0        /* [128][64]  */
#define WHIST_OFF  8192     /* [64][128]  */
#define WFEAT_OFF  16384    /* [64][64]   */
#define WCOMB_OFF  20480    /* [64][128]  */
#define WIH_OFF    28672    /* [512][128] */
#define WHH_OFF    94208    /* [512][128] */
#define W16_TOTAL  159744
#define LOSSBUF_BYTE_OFF 319488  /* = W16_TOTAL*2 */

typedef _Float16 v4h  __attribute__((ext_vector_type(4)));
typedef _Float16 v8h  __attribute__((ext_vector_type(8)));
typedef _Float16 v16h __attribute__((ext_vector_type(16)));
typedef float    v4f  __attribute__((ext_vector_type(4)));
typedef float    v8f  __attribute__((ext_vector_type(8)));

__device__ __forceinline__ v8f vzero8() {
    v8f z;
    #pragma unroll
    for (int i = 0; i < 8; ++i) z[i] = 0.f;
    return z;
}

__device__ __forceinline__ v16h cat16(v8h lo, v8h hi) {
    return __builtin_shufflevector(lo, hi, 0,1,2,3,4,5,6,7,8,9,10,11,12,13,14,15);
}

// A fragment (16x32 f16) from row-major [16][ldk] f16 LDS source.
// Lanes 0-15 -> M=lane, K={kh..+7, kh+16..+23}; kh=(lane>>4)*8.
__device__ __forceinline__ v16h load_a(const _Float16* base, int ldk, int kstep, int lane) {
    int m  = lane & 15;
    int kh = (lane >> 4) * 8;
    const _Float16* p = base + m * ldk + kstep * 32 + kh;
    v8h lo = *(const v8h*)(p);
    v8h hi = *(const v8h*)(p + 16);
    return cat16(lo, hi);
}

// B fragment (32x16 f16): B[k][n] = W[n][k], W row-major [N][K] f16 in global.
__device__ __forceinline__ v16h load_b(const _Float16* W, int ldk, int ntile, int kstep, int lane) {
    int n  = ntile * 16 + (lane & 15);
    int kb = kstep * 32 + ((lane >> 4) * 8);
    const _Float16* p = W + n * ldk + kb;
    v8h lo = *(const v8h*)(p);
    v8h hi = *(const v8h*)(p + 16);
    return cat16(lo, hi);
}

#define WMMA(a, b, c) \
    __builtin_amdgcn_wmma_f32_16x16x32_f16(false, (a), false, (b), (short)0, (c), false, false)

__device__ __forceinline__ float sigmoidf_(float x) {
    return 1.f / (1.f + __expf(-x));
}

// ---------------------------------------------------------------------------
// Kernel 1: convert weights to f16 in workspace (zero W_feat diagonal).
// ---------------------------------------------------------------------------
__global__ void rits_convert(const float* __restrict__ Wdh,
                             const float* __restrict__ Whist,
                             const float* __restrict__ Wfeat,
                             const float* __restrict__ Wcomb,
                             const float* __restrict__ Wih,
                             const float* __restrict__ Whh,
                             _Float16* __restrict__ ws16) {
    int i = blockIdx.x * blockDim.x + threadIdx.x;
    if (i < H_ * F_)  ws16[WDH_OFF   + i] = (_Float16)Wdh[i];
    if (i < F_ * H_)  ws16[WHIST_OFF + i] = (_Float16)Whist[i];
    if (i < F_ * F_) {
        int r = i >> 6, c = i & 63;
        ws16[WFEAT_OFF + i] = (r == c) ? (_Float16)0.f : (_Float16)Wfeat[i];
    }
    if (i < F_ * F2_) ws16[WCOMB_OFF + i] = (_Float16)Wcomb[i];
    if (i < G_ * F2_) ws16[WIH_OFF   + i] = (_Float16)Wih[i];
    if (i < G_ * H_)  ws16[WHH_OFF   + i] = (_Float16)Whh[i];
}

// ---------------------------------------------------------------------------
// Kernel 2: main RITS scan. One block = 16 batch rows, 4 waves (one/SIMD),
// T=64 steps. All weight B-fragments live in registers for the whole scan.
// ---------------------------------------------------------------------------
__global__ __launch_bounds__(NTHREADS, 1)
void rits_main(const float* __restrict__ values,
               const float* __restrict__ masks,
               const float* __restrict__ deltas,
               const float* __restrict__ b_dh,
               const float* __restrict__ W_dx,
               const float* __restrict__ b_dx,
               const float* __restrict__ b_hist,
               const float* __restrict__ b_feat,
               const float* __restrict__ b_comb,
               const float* __restrict__ b_ih,
               const float* __restrict__ b_hh,
               const _Float16* __restrict__ ws16,
               float* __restrict__ out_imp,
               float* __restrict__ out_est,
               float* __restrict__ out_h,
               float* __restrict__ lossbuf) {
    __shared__ __align__(16) float    sx[TILE_M][F_];
    __shared__ __align__(16) float    sm[TILE_M][F_];
    __shared__ __align__(16) _Float16 sd16[TILE_M][F_];
    __shared__ __align__(16) float    sh32[TILE_M][H_];
    __shared__ __align__(16) _Float16 sh16[TILE_M][H_];
    __shared__ __align__(16) _Float16 scomb16[TILE_M][F2_];   // [gamma_x | m]
    __shared__ __align__(16) _Float16 sxc16[TILE_M][F_];
    __shared__ __align__(16) _Float16 sinp16[TILE_M][F2_];    // [c_c | m]
    __shared__ float sloss[4];

    const int tid   = threadIdx.x;
    const int lane  = tid & 31;
    const int w     = tid >> 5;              // wave id 0..3
    const int row0  = blockIdx.x * TILE_M;
    const int cn    = lane & 15;             // column within 16-wide tile
    const int rbase = (lane >> 4) * 8;       // row base for C/D fragment

    // ---- hoist: all weight B-fragments into registers (live across T loop) ----
    v16h Bdh[2][2], Bhist[4], Bfeat[2], Bcomb[4], Bih[2][4][4], Bhh[2][4][4];
    #pragma unroll
    for (int q = 0; q < 2; ++q)
        #pragma unroll
        for (int ks = 0; ks < 2; ++ks)
            Bdh[q][ks] = load_b(ws16 + WDH_OFF, F_, w + 4 * q, ks, lane);
    #pragma unroll
    for (int ks = 0; ks < 4; ++ks) Bhist[ks] = load_b(ws16 + WHIST_OFF, H_, w, ks, lane);
    #pragma unroll
    for (int ks = 0; ks < 2; ++ks) Bfeat[ks] = load_b(ws16 + WFEAT_OFF, F_, w, ks, lane);
    #pragma unroll
    for (int ks = 0; ks < 4; ++ks) Bcomb[ks] = load_b(ws16 + WCOMB_OFF, F2_, w, ks, lane);
    #pragma unroll
    for (int p = 0; p < 2; ++p)
        #pragma unroll
        for (int g = 0; g < 4; ++g)
            #pragma unroll
            for (int ks = 0; ks < 4; ++ks) {
                Bih[p][g][ks] = load_b(ws16 + WIH_OFF, F2_, w + 4 * p + 8 * g, ks, lane);
                Bhh[p][g][ks] = load_b(ws16 + WHH_OFF, H_,  w + 4 * p + 8 * g, ks, lane);
            }

    // ---- hoist: biases (lane-fixed columns) ----
    const int n_f = w * 16 + cn;                 // F-column for phases 3/4
    const float bdh_r[2] = { b_dh[w * 16 + cn], b_dh[(w + 4) * 16 + cn] };
    const float bhist_r = b_hist[n_f];
    const float bfeat_r = b_feat[n_f];
    const float bcomb_r = b_comb[n_f];
    float bg4[2][4];
    #pragma unroll
    for (int p = 0; p < 2; ++p) {
        int ni = (w + 4 * p) * 16 + cn;
        #pragma unroll
        for (int g = 0; g < 4; ++g)
            bg4[p][g] = b_ih[ni + 128 * g] + b_hh[ni + 128 * g];
    }

    // ---- hoist: diagonal decay params for this thread's 8 load elements ----
    float wdxr[2][4], bdxr[2][4];
    #pragma unroll
    for (int k = 0; k < 2; ++k) {
        int i4 = tid + k * NTHREADS;
        int cc = (i4 & 15) * 4;
        #pragma unroll
        for (int e = 0; e < 4; ++e) {
            wdxr[k][e] = W_dx[(cc + e) * F_ + (cc + e)];
            bdxr[k][e] = b_dx[cc + e];
        }
    }

    // zero recurrent state
    for (int i = tid; i < TILE_M * H_; i += NTHREADS)
        sh32[i >> 7][i & 127] = 0.f;
    v8f creg0 = vzero8();                    // c for h-columns [16w, 16w+16)
    v8f creg1 = vzero8();                    // c for h-columns [64+16w, 64+16w+16)
    __syncthreads();

    for (int t = 0; t < T_; ++t) {
        // ---- phase 1: load x/m/d tile (b128), gamma_x, mask halves ----
        if (tid < 4) sloss[tid] = 0.f;
        #pragma unroll
        for (int k = 0; k < 2; ++k) {
            int i4 = tid + k * NTHREADS;
            int r = i4 >> 4, cc = (i4 & 15) * 4;
            const size_t gi = ((size_t)(row0 + r) * T_ + t) * F_ + cc;
            v4f xv = *(const v4f*)(values + gi);
            v4f mv = *(const v4f*)(masks + gi);
            v4f dv = *(const v4f*)(deltas + gi);
            *(v4f*)&sx[r][cc] = xv;
            *(v4f*)&sm[r][cc] = mv;
            v4h dh, gxh, mh;
            #pragma unroll
            for (int e = 0; e < 4; ++e) {
                dh[e] = (_Float16)dv[e];
                float gx = __expf(-fmaxf(dv[e] * wdxr[k][e] + bdxr[k][e], 0.f));
                gxh[e] = (_Float16)gx;
                mh[e]  = (_Float16)mv[e];
            }
            *(v4h*)&sd16[r][cc]          = dh;
            *(v4h*)&scomb16[r][cc]       = gxh;
            *(v4h*)&scomb16[r][cc + F_]  = mh;
            *(v4h*)&sinp16[r][cc + F_]   = mh;
            if (t + 1 < T_) {
                __builtin_prefetch(values + gi + F_, 0, 0);
                __builtin_prefetch(masks  + gi + F_, 0, 0);
                __builtin_prefetch(deltas + gi + F_, 0, 0);
            }
        }
        __syncthreads();

        // ---- phase 2: gamma_h = exp(-relu(d@W_dh^T+b)); h *= gamma_h ----
        #pragma unroll
        for (int q = 0; q < 2; ++q) {
            v8f acc = vzero8();
            #pragma unroll
            for (int ks = 0; ks < 2; ++ks) {
                v16h a = load_a(&sd16[0][0], F_, ks, lane);
                acc = WMMA(a, Bdh[q][ks], acc);
            }
            int n = (w + 4 * q) * 16 + cn;
            float bb = bdh_r[q];
            #pragma unroll
            for (int j = 0; j < 8; ++j) {
                int m = rbase + j;
                float g  = __expf(-fmaxf(acc[j] + bb, 0.f));
                float hd = sh32[m][n] * g;
                sh32[m][n] = hd;
                sh16[m][n] = (_Float16)hd;
            }
        }
        __syncthreads();

        // ---- phase 3: x_h = h@W_hist^T+b; x_c; loss1 ----
        float l1 = 0.f, l2 = 0.f, l3 = 0.f, den = 0.f;
        v8f xh;
        {
            v8f acc = vzero8();
            #pragma unroll
            for (int ks = 0; ks < 4; ++ks) {
                v16h a = load_a(&sh16[0][0], H_, ks, lane);
                acc = WMMA(a, Bhist[ks], acc);
            }
            #pragma unroll
            for (int j = 0; j < 8; ++j) {
                int m = rbase + j;
                float xhv = acc[j] + bhist_r;
                xh[j] = xhv;
                float mv = sm[m][n_f], xv = sx[m][n_f];
                float xc = mv * xv + (1.f - mv) * xhv;
                sxc16[m][n_f] = (_Float16)xc;
                l1  += fabsf(xhv - xv) * mv;
                den += mv;
            }
        }
        __syncthreads();

        // ---- phase 4: z_h, alpha, c_h, outputs, loss2/3 ----
        {
            v8f accz = vzero8();
            #pragma unroll
            for (int ks = 0; ks < 2; ++ks) {
                v16h a = load_a(&sxc16[0][0], F_, ks, lane);
                accz = WMMA(a, Bfeat[ks], accz);
            }
            v8f acca = vzero8();
            #pragma unroll
            for (int ks = 0; ks < 4; ++ks) {
                v16h a = load_a(&scomb16[0][0], F2_, ks, lane);
                acca = WMMA(a, Bcomb[ks], acca);
            }
            #pragma unroll
            for (int j = 0; j < 8; ++j) {
                int m = rbase + j;
                float mv = sm[m][n_f], xv = sx[m][n_f];
                float zh = accz[j] + bfeat_r;
                float al = sigmoidf_(acca[j] + bcomb_r);
                float ch = al * zh + (1.f - al) * xh[j];
                float cc = mv * xv + (1.f - mv) * ch;
                l2 += fabsf(zh - xv) * mv;
                l3 += fabsf(ch - xv) * mv;
                size_t gi = ((size_t)(row0 + m) * T_ + t) * F_ + n_f;
                out_est[gi] = ch;
                out_imp[gi] = cc;
                sinp16[m][n_f] = (_Float16)cc;
            }
        }

        // wave-reduce loss partials, accumulate to block totals
        #pragma unroll
        for (int off = 16; off > 0; off >>= 1) {
            l1  += __shfl_xor(l1,  off, 32);
            l2  += __shfl_xor(l2,  off, 32);
            l3  += __shfl_xor(l3,  off, 32);
            den += __shfl_xor(den, off, 32);
        }
        if (lane == 0) {
            atomicAdd(&sloss[0], l1);
            atomicAdd(&sloss[1], l2);
            atomicAdd(&sloss[2], l3);
            atomicAdd(&sloss[3], den);
        }
        __syncthreads();

        // ---- phase 5: gates = inp@W_ih^T + h@W_hh^T + biases; LSTM ----
        #pragma unroll
        for (int p = 0; p < 2; ++p) {
            v8f accg[4] = { vzero8(), vzero8(), vzero8(), vzero8() };
            #pragma unroll
            for (int ks = 0; ks < 4; ++ks) {
                v16h a = load_a(&sinp16[0][0], F2_, ks, lane);
                #pragma unroll
                for (int g = 0; g < 4; ++g)
                    accg[g] = WMMA(a, Bih[p][g][ks], accg[g]);
            }
            #pragma unroll
            for (int ks = 0; ks < 4; ++ks) {
                v16h a = load_a(&sh16[0][0], H_, ks, lane);
                #pragma unroll
                for (int g = 0; g < 4; ++g)
                    accg[g] = WMMA(a, Bhh[p][g][ks], accg[g]);
            }
            int ni = (w + 4 * p) * 16 + cn;      // h column index (0..127)
            v8f& cr = p ? creg1 : creg0;
            #pragma unroll
            for (int j = 0; j < 8; ++j) {
                float iv = sigmoidf_(accg[0][j] + bg4[p][0]);
                float fv = sigmoidf_(accg[1][j] + bg4[p][1]);
                float gv = tanhf(accg[2][j] + bg4[p][2]);
                float ov = sigmoidf_(accg[3][j] + bg4[p][3]);
                float cv = fv * cr[j] + iv * gv;
                cr[j] = cv;
                sh32[rbase + j][ni] = ov * tanhf(cv);
            }
        }
        __syncthreads();

        if (tid == 0) {
            float* lb = lossbuf + ((size_t)t * NBLK + blockIdx.x) * 4;
            lb[0] = sloss[0]; lb[1] = sloss[1]; lb[2] = sloss[2]; lb[3] = sloss[3];
        }
        __syncthreads();
    }

    // final hidden state
    for (int i = tid; i < TILE_M * H_; i += NTHREADS) {
        int r = i >> 7, cH = i & 127;
        out_h[(size_t)(row0 + r) * H_ + cH] = sh32[r][cH];
    }
}

// ---------------------------------------------------------------------------
// Kernel 3: final loss reduction across blocks and timesteps.
// ---------------------------------------------------------------------------
__global__ void rits_loss(const float* __restrict__ lossbuf, float* __restrict__ out_loss) {
    __shared__ float red[T_];
    int t = threadIdx.x;  // 0..63
    float n0 = 0.f, n1 = 0.f, n2 = 0.f, dd = 0.f;
    for (int blk = 0; blk < NBLK; ++blk) {
        const float* lb = lossbuf + ((size_t)t * NBLK + blk) * 4;
        n0 += lb[0]; n1 += lb[1]; n2 += lb[2]; dd += lb[3];
    }
    red[t] = (n0 + n1 + n2) / (dd + EPSF);
    __syncthreads();
    if (t == 0) {
        float s = 0.f;
        for (int i = 0; i < T_; ++i) s += red[i];
        out_loss[0] = s / (float)(T_ * 3);
    }
}

// ---------------------------------------------------------------------------
extern "C" void kernel_launch(void* const* d_in, const int* in_sizes, int n_in,
                              void* d_out, int out_size, void* d_ws, size_t ws_size,
                              hipStream_t stream) {
    const float* values = (const float*)d_in[0];
    const float* masks  = (const float*)d_in[1];
    const float* deltas = (const float*)d_in[2];
    const float* W_dh   = (const float*)d_in[3];
    const float* b_dh   = (const float*)d_in[4];
    const float* W_dx   = (const float*)d_in[5];
    const float* b_dx   = (const float*)d_in[6];
    const float* W_hist = (const float*)d_in[7];
    const float* b_hist = (const float*)d_in[8];
    const float* W_feat = (const float*)d_in[9];
    const float* b_feat = (const float*)d_in[10];
    const float* W_comb = (const float*)d_in[11];
    const float* b_comb = (const float*)d_in[12];
    const float* W_ih   = (const float*)d_in[13];
    const float* b_ih   = (const float*)d_in[14];
    const float* W_hh   = (const float*)d_in[15];
    const float* b_hh   = (const float*)d_in[16];

    _Float16* ws16 = (_Float16*)d_ws;
    float* lossbuf = (float*)((char*)d_ws + LOSSBUF_BYTE_OFF);

    float* out_imp  = (float*)d_out;
    float* out_est  = out_imp + (size_t)B_ * T_ * F_;
    float* out_h    = out_est + (size_t)B_ * T_ * F_;
    float* out_loss = out_h + (size_t)B_ * H_;

    rits_convert<<<256, 256, 0, stream>>>(W_dh, W_hist, W_feat, W_comb, W_ih, W_hh, ws16);

    rits_main<<<NBLK, NTHREADS, 0, stream>>>(
        values, masks, deltas,
        b_dh, W_dx, b_dx, b_hist, b_feat, b_comb, b_ih, b_hh,
        ws16, out_imp, out_est, out_h, lossbuf);

    rits_loss<<<1, T_, 0, stream>>>(lossbuf, out_loss);
}
